// TriangleMultiplicativeModule_1726576856770
// MI455X (gfx1250) — matrix operand order repaired and checked by reading, not compiled, not run on hardware
//
#include <hip/hip_runtime.h>

typedef unsigned int u32;
typedef __bf16 bf16;
typedef bf16 bf16x16 __attribute__((ext_vector_type(16)));
typedef float f32x8  __attribute__((ext_vector_type(8)));
typedef u32  u32x4   __attribute__((ext_vector_type(4)));

#define NN 384
#define DD 128
#define EPSLN 1e-5f

__device__ __forceinline__ unsigned short f2bf(float f) {
    union { float f; u32 u; } c; c.f = f;
    u32 u = c.u + 0x7FFFu + ((c.u >> 16) & 1u);   // round-to-nearest-even
    return (unsigned short)(u >> 16);
}
__device__ __forceinline__ float bf2f(unsigned short h) {
    union { u32 u; float f; } c; c.u = ((u32)h) << 16; return c.f;
}
__device__ __forceinline__ float sigm(float x) { return 1.f / (1.f + __expf(-x)); }

union Frag { bf16x16 v; u32 u[8]; u32x4 q[2]; };

// A-fragment (16x32 bf16): lane holds row M=lane%16.
// half = lane>>4:  u[0..3] = k{0..7|8..15} pairs, u[4..7] = k{16..23|24..31}.
__device__ __forceinline__ void loadA(Frag& f, const u32* row, int kq, int half) {
    const u32* b = row + kq + half * 4;
    f.q[0] = *reinterpret_cast<const u32x4*>(b);
    f.q[1] = *reinterpret_cast<const u32x4*>(b + 8);
}
// B-fragment (32x16 bf16): lane holds column N=lane%16; same k layout as A: u[0..3] = k{0..7|8..15}, u[4..7] = k{16..23|24..31}.
__device__ __forceinline__ void loadB(Frag& f, const u32* colrow, int kq, int half) {
    const u32* b = colrow + kq + half * 4;
    f.q[0] = *reinterpret_cast<const u32x4*>(b);
    f.q[1] = *reinterpret_cast<const u32x4*>(b + 8);
}

// ---------------- kernel 1: weights -> bf16, transposed [N][K] ----------------
__global__ __launch_bounds__(256) void k_prep(
    const float* __restrict__ w0, const float* __restrict__ w1,
    const float* __restrict__ w2, const float* __restrict__ w3,
    const float* __restrict__ w4, const float* __restrict__ w5,
    unsigned short* __restrict__ wt)
{
    int idx = blockIdx.x * 256 + threadIdx.x;          // 6*128*128 = 98304
    int m = idx >> 14;
    int rem = idx & 16383;
    int k = rem >> 7, n = rem & 127;
    const float* src = (m == 0) ? w0 : (m == 1) ? w1 : (m == 2) ? w2
                     : (m == 3) ? w3 : (m == 4) ? w4 : w5;
    wt[m * 16384 + n * 128 + k] = f2bf(src[k * 128 + n]);
}

// ------------- kernel 2: LN + 5 projections + sigmoid gating (WMMA) -----------
__global__ __launch_bounds__(256) void k_proj(
    const float* __restrict__ x,
    const float* __restrict__ ln_g, const float* __restrict__ ln_b,
    const u32* __restrict__ wt,     // 6 mats of 8192 dwords, [n][k] bf16
    const float* __restrict__ bL,  const float* __restrict__ bR,
    const float* __restrict__ bLG, const float* __restrict__ bRG,
    const float* __restrict__ bOG,
    unsigned short* __restrict__ leftW,
    unsigned short* __restrict__ rightW,
    unsigned short* __restrict__ ogateW)
{
    __shared__ unsigned short xn[256][128];            // 64 KB bf16 LN output
    const int tid = threadIdx.x;
    const int p0 = blockIdx.x * 256;

    // ---- LayerNorm over dim, one thread per row ----
    {
        const float* xr = x + (size_t)(p0 + tid) * 128;
        float s = 0.f;
        for (int e = 0; e < 128; ++e) s += xr[e];
        float mu = s * (1.f / 128.f);
        float var = 0.f;
        for (int e = 0; e < 128; ++e) {
            float d = xr[e] - mu;
            var += d * d;
            xn[tid][e] = f2bf(d);
        }
        float rs = rsqrtf(var * (1.f / 128.f) + EPSLN);
        for (int e = 0; e < 128; ++e)
            xn[tid][e] = f2bf(bf2f(xn[tid][e]) * rs * ln_g[e] + ln_b[e]);
    }
    __syncthreads();

    const int wv = tid >> 5, lane = tid & 31;
    const int half = lane >> 4, lm = lane & 15;

    for (int sub = 0; sub < 2; ++sub) {
        const int m0 = (wv * 2 + sub) * 16;
        Frag A[4];
        const u32* arow = reinterpret_cast<const u32*>(&xn[m0 + lm][0]);
#pragma unroll
        for (int kt = 0; kt < 4; ++kt) loadA(A[kt], arow, kt * 16, half);

        for (int nt = 0; nt < 8; ++nt) {
            const int col = nt * 16 + lm;
            // preload all 4 B-fragments before the WMMA chain so the
            // global_load_b128 clause overlaps the matrix pipe
            auto gemm = [&](const u32* wmat, const float* bias) -> f32x8 {
                float bv = bias[col];
                f32x8 acc = {bv, bv, bv, bv, bv, bv, bv, bv};
                const u32* brow = wmat + col * 64;
                Frag B[4];
#pragma unroll
                for (int kt = 0; kt < 4; ++kt) loadB(B[kt], brow, kt * 16, half);
#pragma unroll
                for (int kt = 0; kt < 4; ++kt)
                    acc = __builtin_amdgcn_wmma_f32_16x16x32_bf16(
                        false, A[kt].v, false, B[kt].v, (short)0, acc, false, false);
                return acc;
            };
            f32x8 L  = gemm(wt + 0 * 8192, bL);
            f32x8 G  = gemm(wt + 2 * 8192, bLG);
            f32x8 R  = gemm(wt + 1 * 8192, bR);
            f32x8 RG = gemm(wt + 3 * 8192, bRG);
            f32x8 O  = gemm(wt + 4 * 8192, bOG);
#pragma unroll
            for (int r = 0; r < 8; ++r) {
                int m  = r + half * 8;
                int p  = p0 + m0 + m;
                int bb = p / (NN * NN);
                int rem = p - bb * (NN * NN);
                int kk = rem / NN;
                int cc = rem - kk * NN;
                int tIdx = ((bb * NN + cc) * 128 + col) * NN + kk;
                leftW[tIdx]  = f2bf(L[r] * sigm(G[r]));
                rightW[tIdx] = f2bf(R[r] * sigm(RG[r]));
                ogateW[p * 128 + col] = f2bf(sigm(O[r]));
            }
        }
    }
}

// ------- kernel 3: triangle einsum (WMMA) + out-LN + gate + out-proj (WMMA) ---
__global__ __launch_bounds__(256) void k_tri(
    const unsigned short* __restrict__ leftW,   // [b][j][d][k]
    const unsigned short* __restrict__ rightW,  // [b][i][d][k]
    const unsigned short* __restrict__ ogateW,  // [p][d]
    const float* __restrict__ on_g, const float* __restrict__ on_b,
    const u32* __restrict__ wtO,                // to_out bf16 [n][k]
    const float* __restrict__ bO,
    float* __restrict__ out)
{
    __shared__ unsigned short tri[256][128];    // [(il*16+jl)][d] bf16, 64 KB
    const int tid = threadIdx.x, wv = tid >> 5, lane = tid & 31;
    const int half = lane >> 4, lm = lane & 15;
    const int j0 = blockIdx.x * 16, i0 = blockIdx.y * 16, bb = blockIdx.z;

    const u32* rW = reinterpret_cast<const u32*>(rightW);
    const u32* lW = reinterpret_cast<const u32*>(leftW);
    const int arowBase = (bb * NN + i0 + lm) * 128;   // + d, then *192 dwords
    const int browBase = (bb * NN + j0 + lm) * 128;

    // out[i,j,d] = sum_k right[k,i,d] * left[k,j,d]  (per-d 16x16x384 GEMM)
    for (int dc = 0; dc < 2; ++dc) {
        const int dbase = wv * 16 + dc * 8;
        f32x8 acc[8];
#pragma unroll
        for (int dd = 0; dd < 8; ++dd) acc[dd] = f32x8{0, 0, 0, 0, 0, 0, 0, 0};

        // double-buffered fragment pipeline over (kt, dd)
        Frag Af[2], Bf[2];
        loadA(Af[0], rW + (arowBase + dbase) * 192, 0, half);
        loadB(Bf[0], lW + (browBase + dbase) * 192, 0, half);
        for (int kt = 0; kt < 12; ++kt) {
            const int kq = kt * 16;
#pragma unroll
            for (int dd = 0; dd < 8; ++dd) {
                const int cur = dd & 1, nxt = cur ^ 1;
                // issue next channel's (A,B) loads before consuming current
                if (dd < 7) {
                    const int dn = dbase + dd + 1;
                    loadA(Af[nxt], rW + (arowBase + dn) * 192, kq, half);
                    loadB(Bf[nxt], lW + (browBase + dn) * 192, kq, half);
                } else if (kt < 11) {
                    loadA(Af[nxt], rW + (arowBase + dbase) * 192, kq + 16, half);
                    loadB(Bf[nxt], lW + (browBase + dbase) * 192, kq + 16, half);
                }
                acc[dd] = __builtin_amdgcn_wmma_f32_16x16x32_bf16(
                    false, Af[cur].v, false, Bf[cur].v, (short)0, acc[dd], false, false);
            }
        }
#pragma unroll
        for (int dd = 0; dd < 8; ++dd)
#pragma unroll
            for (int r = 0; r < 8; ++r)
                tri[(r + half * 8) * 16 + lm][dbase + dd] = f2bf(acc[dd][r]);
    }
    __syncthreads();

    // ---- output LayerNorm over d + sigmoid gate, in place (one thread/(i,j)) ----
    {
        const int il = tid >> 4, jl = tid & 15;
        unsigned short* row = tri[tid];
        float s = 0.f;
        for (int e = 0; e < 128; ++e) s += bf2f(row[e]);
        float mu = s * (1.f / 128.f);
        float var = 0.f;
        for (int e = 0; e < 128; ++e) { float d = bf2f(row[e]) - mu; var += d * d; }
        float rs = rsqrtf(var * (1.f / 128.f) + EPSLN);
        const int p = (bb * NN + i0 + il) * NN + (j0 + jl);
        const unsigned short* og = ogateW + p * 128;
        for (int e = 0; e < 128; ++e) {
            float d = (bf2f(row[e]) - mu) * rs * on_g[e] + on_b[e];
            row[e] = f2bf(d * bf2f(og[e]));
        }
    }
    __syncthreads();

    // ---- out projection: (256 x 128) @ (128 x 128) via WMMA ----
    for (int sub = 0; sub < 2; ++sub) {
        const int m0 = (wv * 2 + sub) * 16;
        Frag A[4];
        const u32* arow = reinterpret_cast<const u32*>(&tri[m0 + lm][0]);
#pragma unroll
        for (int kt = 0; kt < 4; ++kt) loadA(A[kt], arow, kt * 16, half);
        for (int nt = 0; nt < 8; ++nt) {
            const int col = nt * 16 + lm;
            float bv = bO[col];
            f32x8 acc = {bv, bv, bv, bv, bv, bv, bv, bv};
            const u32* brow = wtO + col * 64;
            Frag B[4];
#pragma unroll
            for (int kt = 0; kt < 4; ++kt) loadB(B[kt], brow, kt * 16, half);
#pragma unroll
            for (int kt = 0; kt < 4; ++kt)
                acc = __builtin_amdgcn_wmma_f32_16x16x32_bf16(
                    false, A[kt].v, false, B[kt].v, (short)0, acc, false, false);
#pragma unroll
            for (int r = 0; r < 8; ++r) {
                int m = m0 + r + half * 8;
                int il = m >> 4, jl = m & 15;
                size_t op = ((size_t)(bb * NN + i0 + il) * NN + (j0 + jl)) * 128 + col;
                out[op] = acc[r];
            }
        }
    }
}

extern "C" void kernel_launch(void* const* d_in, const int* in_sizes, int n_in,
                              void* d_out, int out_size, void* d_ws, size_t ws_size,
                              hipStream_t stream) {
    (void)in_sizes; (void)n_in; (void)out_size; (void)ws_size;
    const float* x    = (const float*)d_in[0];
    const float* ng   = (const float*)d_in[1];
    const float* nb   = (const float*)d_in[2];
    const float* lw   = (const float*)d_in[3];
    const float* lb   = (const float*)d_in[4];
    const float* rw   = (const float*)d_in[5];
    const float* rb   = (const float*)d_in[6];
    const float* lgw  = (const float*)d_in[7];
    const float* lgb  = (const float*)d_in[8];
    const float* rgw  = (const float*)d_in[9];
    const float* rgb  = (const float*)d_in[10];
    const float* ogw  = (const float*)d_in[11];
    const float* ogb  = (const float*)d_in[12];
    const float* ong  = (const float*)d_in[13];
    const float* onb  = (const float*)d_in[14];
    const float* tow  = (const float*)d_in[15];
    const float* tob  = (const float*)d_in[16];

    char* ws = (char*)d_ws;
    unsigned short* wt     = (unsigned short*)ws;                 // 6*16384 bf16
    unsigned short* leftW  = (unsigned short*)(ws + 196608);      // 294912*128 bf16
    unsigned short* rightW = leftW  + (size_t)294912 * 128;
    unsigned short* ogateW = rightW + (size_t)294912 * 128;

    k_prep<<<384, 256, 0, stream>>>(lw, rw, lgw, rgw, ogw, tow, wt);
    k_proj<<<1152, 256, 0, stream>>>(x, ng, nb, (const u32*)wt,
                                     lb, rb, lgb, rgb, ogb,
                                     leftW, rightW, ogateW);
    k_tri<<<dim3(24, 24, 2), 256, 0, stream>>>(leftW, rightW, ogateW,
                                               ong, onb,
                                               (const u32*)(wt + 5 * 16384), tob,
                                               (float*)d_out);
}
